// FuXiTransformerStage_79577154060524
// MI455X (gfx1250) — matrix operand order repaired and checked by reading, not compiled, not run on hardware
//
#include <hip/hip_runtime.h>
#include <hip/hip_bf16.h>
#include <math.h>

typedef __attribute__((ext_vector_type(16))) _Float16 v16h;
typedef __attribute__((ext_vector_type(8)))  float    v8f;

#define TOK   32768   // total tokens = 4*64*128
#define CC    768
#define NQKV  2304
#define NHID  3072
#define NHEAD 8
#define HD    96
#define NWIN  512

// ---------------- async global->LDS copy (gfx1250 ASYNCcnt path) -------------
#if defined(__gfx1250__) && __has_builtin(__builtin_amdgcn_global_load_async_to_lds_b128)
#define HAVE_ASYNC 1
typedef int v4i_async __attribute__((vector_size(16)));
typedef __attribute__((address_space(1))) v4i_async* g_v4i_p;
typedef __attribute__((address_space(3))) v4i_async* l_v4i_p;
#else
#define HAVE_ASYNC 0
#endif

__device__ __forceinline__ void async_cp16(_Float16* lds, const _Float16* g) {
#if HAVE_ASYNC
  __builtin_amdgcn_global_load_async_to_lds_b128(
      (g_v4i_p)(v4i_async*)(void*)g, (l_v4i_p)(v4i_async*)(void*)lds, 0, 0);
#else
  uint32_t* d = (uint32_t*)lds;
  const uint32_t* s = (const uint32_t*)g;
  d[0] = s[0]; d[1] = s[1]; d[2] = s[2]; d[3] = s[3];
#endif
}
__device__ __forceinline__ void wait_async() {
#if HAVE_ASYNC
#if __has_builtin(__builtin_amdgcn_s_wait_asynccnt)
  __builtin_amdgcn_s_wait_asynccnt(0);
#else
  asm volatile("s_wait_asynccnt 0x0" ::: "memory");
#endif
#endif
}

// window/token index -> image pixel index (handles cyclic shift both ways)
__device__ __forceinline__ int win_to_img(int m, int shift) {
  int w  = m >> 6;
  int n  = m & 63;
  int b  = w >> 7;           // 128 windows per image (8 x 16)
  int wl = w & 127;
  int wh = wl >> 4;
  int ww = wl & 15;
  int i  = n >> 3, j = n & 7;
  int h  = ((wh << 3) + i + shift) & 63;
  int wi = ((ww << 3) + j + shift) & 127;
  return ((b << 6) + h) * 128 + wi;
}

// ---- WMMA fragment helpers (contiguous k = 16*half + s in both A and B) ----------------------
__device__ __forceinline__ v16h load_a_frag(const _Float16* base, int ld, int row, int half) {
  v16h a;
  const _Float16* p0 = base + row * ld + 16 * half;
  const _Float16* p1 = p0 + 8;
#pragma unroll
  for (int j = 0; j < 8; ++j) { a[j] = p0[j]; a[8 + j] = p1[j]; }
  return a;
}
__device__ __forceinline__ v16h load_b_frag(const _Float16* base, int ld, int col, int half) {
  v16h b;
  const _Float16* p = base + col * ld + 16 * half;
#pragma unroll
  for (int j = 0; j < 16; ++j) b[j] = p[j];
  return b;
}
__device__ __forceinline__ v8f wmma_f16(v16h a, v16h b, v8f c) {
  return __builtin_amdgcn_wmma_f32_16x16x32_f16(false, a, false, b, (short)0, c, false, false);
}

// ---- fp32 -> f16 flat convert (weights, once per depth) ---------------------
__global__ __launch_bounds__(256) void k_cvt16(const float* __restrict__ s,
                                               _Float16* __restrict__ d, int n) {
  int i = blockIdx.x * 256 + threadIdx.x;
  if (i < n) d[i] = (_Float16)s[i];
}

// ---- roll + window partition + fp32->f16 ------------------------------------
__global__ __launch_bounds__(256) void k_partition(const float* __restrict__ x,
                                                   _Float16* __restrict__ xw, int shift) {
  long idx = (long)blockIdx.x * 256 + threadIdx.x;
  int m = (int)(idx / CC);
  int c = (int)(idx - (long)m * CC);
  int src = win_to_img(m, shift);
  xw[idx] = (_Float16)x[(size_t)src * CC + c];
}

// ---- qkv bias = [q_bias, 0, v_bias] -----------------------------------------
__global__ void k_qkv_bias(const float* __restrict__ qb, const float* __restrict__ vb,
                           float* __restrict__ out) {
  int c = blockIdx.x * 256 + threadIdx.x;
  if (c >= NQKV) return;
  float v = 0.f;
  if (c < CC)            v = qb[c];
  else if (c >= 2 * CC)  v = vb[c - 2 * CC];
  out[c] = v;
}

// ---- continuous relative-position-bias MLP table (225 x 8) ------------------
__global__ void k_cpb(const float* __restrict__ w1, const float* __restrict__ b1,
                      const float* __restrict__ w2, float* __restrict__ rpb) {
  int t = threadIdx.x;
  if (t >= 225) return;
  int a = t / 15, bq = t % 15;
  float c0 = (float)(a - 7) / 7.0f * 8.0f;
  float c1 = (float)(bq - 7) / 7.0f * 8.0f;
  const float inv_l8 = 1.0f / 3.0f;  // 1/log2(8)
  float s0 = (c0 > 0.f) ? 1.f : ((c0 < 0.f) ? -1.f : 0.f);
  float s1 = (c1 > 0.f) ? 1.f : ((c1 < 0.f) ? -1.f : 0.f);
  c0 = s0 * log2f(fabsf(c0) + 1.f) * inv_l8;
  c1 = s1 * log2f(fabsf(c1) + 1.f) * inv_l8;
  float acc[NHEAD] = {};
  for (int k = 0; k < 512; ++k) {
    float h = fmaxf(w1[2 * k] * c0 + w1[2 * k + 1] * c1 + b1[k], 0.f);
#pragma unroll
    for (int hh = 0; hh < NHEAD; ++hh) acc[hh] += w2[hh * 512 + k] * h;
  }
#pragma unroll
  for (int hh = 0; hh < NHEAD; ++hh)
    rpb[t * NHEAD + hh] = 16.f / (1.f + expf(-acc[hh]));
}

// ---- tiled WMMA GEMM: out = A[f16 MxK] * W[f16 NxK]^T + bias ----------------
// Block tile 128x64, wave tile 32x32, double-buffered async LDS staging.
// outh != nullptr -> f16 output; else f32 output with optional window->image remap
__global__ __launch_bounds__(256) void k_gemm(const _Float16* __restrict__ A,
                                              const _Float16* __restrict__ W,
                                              const float* __restrict__ bias,
                                              float* __restrict__ outf,
                                              _Float16* __restrict__ outh,
                                              int M, int N, int K, int map_shift) {
  __shared__ _Float16 As[2 * 128 * 40];
  __shared__ _Float16 Bs[2 * 64 * 40];
  const int tid = threadIdx.x, lane = tid & 31, wv = tid >> 5;
  const int wr = wv >> 1, wc = wv & 1, half = lane >> 4, lrow = lane & 15;
  const int m0 = blockIdx.y * 128, n0 = blockIdx.x * 64;

  auto stage = [&](int k0, int bi) {
    {  // A: 128 rows x 32 halves = 512 x 16B chunks, 2 per thread
      int c = tid;
      int row = c >> 2, seg = c & 3;
      async_cp16(&As[bi * 5120 + row * 40 + seg * 8], A + (size_t)(m0 + row) * K + k0 + seg * 8);
      c = tid + 256;
      row = c >> 2; seg = c & 3;
      async_cp16(&As[bi * 5120 + row * 40 + seg * 8], A + (size_t)(m0 + row) * K + k0 + seg * 8);
    }
    {  // B: 64 rows x 32 halves = 256 x 16B chunks, 1 per thread
      int row = tid >> 2, seg = tid & 3;
      async_cp16(&Bs[bi * 2560 + row * 40 + seg * 8], W + (size_t)(n0 + row) * K + k0 + seg * 8);
    }
  };

  v8f c00 = {}, c01 = {}, c10 = {}, c11 = {};
  stage(0, 0);
  wait_async();
  __syncthreads();
  for (int k0 = 0; k0 < K; k0 += 32) {
    const int cur = (k0 >> 5) & 1;
    if (k0 + 32 < K) {
      stage(k0 + 32, cur ^ 1);
      __builtin_prefetch(A + (size_t)(m0 + (tid >> 1)) * K + k0 + 64, 0, 1);  // global_prefetch_b8
    }
    const _Float16* Ab = &As[cur * 5120];
    const _Float16* Bb = &Bs[cur * 2560];
    v16h a0 = load_a_frag(Ab, 40, wr * 32 + lrow, half);
    v16h a1 = load_a_frag(Ab, 40, wr * 32 + 16 + lrow, half);
    v16h b0 = load_b_frag(Bb, 40, wc * 32 + lrow, half);
    v16h b1 = load_b_frag(Bb, 40, wc * 32 + 16 + lrow, half);
    c00 = wmma_f16(a0, b0, c00);
    c01 = wmma_f16(a0, b1, c01);
    c10 = wmma_f16(a1, b0, c10);
    c11 = wmma_f16(a1, b1, c11);
    wait_async();
    __syncthreads();
  }
#pragma unroll
  for (int r = 0; r < 8; ++r) {
    int mr0 = m0 + wr * 32 + r + 8 * half;
    int mr1 = mr0 + 16;
    int nc0 = n0 + wc * 32 + lrow, nc1 = nc0 + 16;
    float v00 = c00[r] + bias[nc0], v01 = c01[r] + bias[nc1];
    float v10 = c10[r] + bias[nc0], v11 = c11[r] + bias[nc1];
    if (outh) {
      outh[(size_t)mr0 * N + nc0] = (_Float16)v00;
      outh[(size_t)mr0 * N + nc1] = (_Float16)v01;
      outh[(size_t)mr1 * N + nc0] = (_Float16)v10;
      outh[(size_t)mr1 * N + nc1] = (_Float16)v11;
    } else {
      int o0 = (map_shift >= 0) ? win_to_img(mr0, map_shift) : mr0;
      int o1 = (map_shift >= 0) ? win_to_img(mr1, map_shift) : mr1;
      outf[(size_t)o0 * N + nc0] = v00;
      outf[(size_t)o0 * N + nc1] = v01;
      outf[(size_t)o1 * N + nc0] = v10;
      outf[(size_t)o1 * N + nc1] = v11;
    }
  }
}

// ---- fused gated-GELU fc1 GEMM: out = gelu(A*Wg^T+bg) * (A*Wv^T+bv) ---------
__global__ __launch_bounds__(256) void k_gemm_glu(const _Float16* __restrict__ A,
                                                  const _Float16* __restrict__ W,  // [2*Nh][K] f16
                                                  const float* __restrict__ bias,
                                                  _Float16* __restrict__ outh,
                                                  int M, int Nh, int K) {
  __shared__ _Float16 As[2 * 128 * 40];
  __shared__ _Float16 Bg[2 * 64 * 40];
  __shared__ _Float16 Bv[2 * 64 * 40];
  const int tid = threadIdx.x, lane = tid & 31, wv = tid >> 5;
  const int wr = wv >> 1, wc = wv & 1, half = lane >> 4, lrow = lane & 15;
  const int m0 = blockIdx.y * 128, n0 = blockIdx.x * 64;

  auto stage = [&](int k0, int bi) {
    int c = tid, row = c >> 2, seg = c & 3;
    async_cp16(&As[bi * 5120 + row * 40 + seg * 8], A + (size_t)(m0 + row) * K + k0 + seg * 8);
    c = tid + 256; row = c >> 2; seg = c & 3;
    async_cp16(&As[bi * 5120 + row * 40 + seg * 8], A + (size_t)(m0 + row) * K + k0 + seg * 8);
    row = tid >> 2; seg = tid & 3;
    async_cp16(&Bg[bi * 2560 + row * 40 + seg * 8], W + (size_t)(n0 + row) * K + k0 + seg * 8);
    async_cp16(&Bv[bi * 2560 + row * 40 + seg * 8], W + (size_t)(Nh + n0 + row) * K + k0 + seg * 8);
  };

  v8f g00 = {}, g01 = {}, g10 = {}, g11 = {};
  v8f x00 = {}, x01 = {}, x10 = {}, x11 = {};
  stage(0, 0);
  wait_async();
  __syncthreads();
  for (int k0 = 0; k0 < K; k0 += 32) {
    const int cur = (k0 >> 5) & 1;
    if (k0 + 32 < K) stage(k0 + 32, cur ^ 1);
    const _Float16* Ab  = &As[cur * 5120];
    const _Float16* Bgb = &Bg[cur * 2560];
    const _Float16* Bvb = &Bv[cur * 2560];
    v16h a0  = load_a_frag(Ab, 40, wr * 32 + lrow, half);
    v16h a1  = load_a_frag(Ab, 40, wr * 32 + 16 + lrow, half);
    v16h bg0 = load_b_frag(Bgb, 40, wc * 32 + lrow, half);
    v16h bg1 = load_b_frag(Bgb, 40, wc * 32 + 16 + lrow, half);
    v16h bv0 = load_b_frag(Bvb, 40, wc * 32 + lrow, half);
    v16h bv1 = load_b_frag(Bvb, 40, wc * 32 + 16 + lrow, half);
    g00 = wmma_f16(a0, bg0, g00); g01 = wmma_f16(a0, bg1, g01);
    g10 = wmma_f16(a1, bg0, g10); g11 = wmma_f16(a1, bg1, g11);
    x00 = wmma_f16(a0, bv0, x00); x01 = wmma_f16(a0, bv1, x01);
    x10 = wmma_f16(a1, bv0, x10); x11 = wmma_f16(a1, bv1, x11);
    wait_async();
    __syncthreads();
  }
#pragma unroll
  for (int r = 0; r < 8; ++r) {
    int mr0 = m0 + wr * 32 + r + 8 * half;
    int mr1 = mr0 + 16;
    int nc0 = n0 + wc * 32 + lrow, nc1 = nc0 + 16;
    float b0 = bias[nc0], b1 = bias[nc1], vb0 = bias[Nh + nc0], vb1 = bias[Nh + nc1];
    float gg, vv;
    gg = g00[r] + b0; vv = x00[r] + vb0;
    outh[(size_t)mr0 * Nh + nc0] = (_Float16)(0.5f * gg * (1.f + erff(gg * 0.70710678f)) * vv);
    gg = g01[r] + b1; vv = x01[r] + vb1;
    outh[(size_t)mr0 * Nh + nc1] = (_Float16)(0.5f * gg * (1.f + erff(gg * 0.70710678f)) * vv);
    gg = g10[r] + b0; vv = x10[r] + vb0;
    outh[(size_t)mr1 * Nh + nc0] = (_Float16)(0.5f * gg * (1.f + erff(gg * 0.70710678f)) * vv);
    gg = g11[r] + b1; vv = x11[r] + vb1;
    outh[(size_t)mr1 * Nh + nc1] = (_Float16)(0.5f * gg * (1.f + erff(gg * 0.70710678f)) * vv);
  }
}

// ---- cosine window attention (one block per (window, head)) -----------------
__global__ __launch_bounds__(256) void k_attn(const _Float16* __restrict__ qkv,
                                              const float* __restrict__ rpb,
                                              const float* __restrict__ lsc,
                                              _Float16* __restrict__ out, int shift) {
  __shared__ _Float16 qs[64 * 104];
  __shared__ _Float16 ks[64 * 104];
  __shared__ _Float16 vs[64 * 104];
  __shared__ float    sa[64 * 68];
  __shared__ _Float16 ah[64 * 72];
  const int tid = threadIdx.x, lane = tid & 31, wv = tid >> 5;
  const int half = lane >> 4, lrow = lane & 15;
  const int wid = blockIdx.x >> 3, head = blockIdx.x & 7;
  {  // async-stage q/k/v: 64 rows x 96 halves = 768 x 16B chunks per matrix
#pragma unroll
    for (int t = 0; t < 3; ++t) {
      int c = tid + t * 256;
      int row = c / 12, seg = c % 12;
      const _Float16* gq = qkv + (size_t)(wid * 64 + row) * NQKV + head * HD + seg * 8;
      int loff = row * 104 + seg * 8;
      async_cp16(&qs[loff], gq);
      async_cp16(&ks[loff], gq + CC);
      async_cp16(&vs[loff], gq + 2 * CC);
    }
    wait_async();
  }
  __syncthreads();
  if (tid < 128) {  // L2-normalize q,k rows
    _Float16* rp = (tid < 64) ? (qs + tid * 104) : (ks + (tid - 64) * 104);
    float s = 0.f;
    for (int d = 0; d < HD; ++d) { float v = (float)rp[d]; s += v * v; }
    float inv = 1.f / fmaxf(sqrtf(s), 1e-12f);
    for (int d = 0; d < HD; ++d) rp[d] = (_Float16)((float)rp[d] * inv);
  }
  __syncthreads();
  {  // attn = qn @ kn^T (64x64, K=96)
    const int wr = wv >> 1, wc = wv & 1;
    v8f c0 = {}, c1 = {};
#pragma unroll
    for (int k0 = 0; k0 < HD; k0 += 32) {
      v16h a  = load_a_frag(qs + k0, 104, wr * 16 + lrow, half);
      v16h b0 = load_b_frag(ks + k0, 104, wc * 32 + lrow, half);
      v16h b1 = load_b_frag(ks + k0, 104, wc * 32 + 16 + lrow, half);
      c0 = wmma_f16(a, b0, c0);
      c1 = wmma_f16(a, b1, c1);
    }
    float scale = expf(fminf(lsc[head], 4.6051702f));  // log(100)
    int wl = wid & 127, wh = wl >> 4, ww = wl & 15;
#pragma unroll
    for (int r = 0; r < 8; ++r) {
      int n = wr * 16 + r + 8 * half;
#pragma unroll
      for (int s = 0; s < 2; ++s) {
        int m = wc * 32 + s * 16 + lrow;
        float v = (s == 0 ? c0[r] : c1[r]) * scale;
        int i1 = n >> 3, j1 = n & 7, i2 = m >> 3, j2 = m & 7;
        v += rpb[((i1 - i2 + 7) * 15 + (j1 - j2 + 7)) * NHEAD + head];
        if (shift > 0) {
          int h1 = wh * 8 + i1, x1 = ww * 8 + j1;
          int h2 = wh * 8 + i2, x2 = ww * 8 + j2;
          int r1 = ((h1 < 56) ? 0 : (h1 < 60) ? 1 : 2) * 3 + ((x1 < 120) ? 0 : (x1 < 124) ? 1 : 2);
          int r2 = ((h2 < 56) ? 0 : (h2 < 60) ? 1 : 2) * 3 + ((x2 < 120) ? 0 : (x2 < 124) ? 1 : 2);
          if (r1 != r2) v -= 100.f;
        }
        sa[n * 68 + m] = v;
      }
    }
  }
  __syncthreads();
  if (tid < 64) {  // row softmax
    float mx = -1e30f;
    for (int m = 0; m < 64; ++m) mx = fmaxf(mx, sa[tid * 68 + m]);
    float sum = 0.f;
    for (int m = 0; m < 64; ++m) { float e = expf(sa[tid * 68 + m] - mx); sa[tid * 68 + m] = e; sum += e; }
    float inv = 1.f / sum;
    for (int m = 0; m < 64; ++m) ah[tid * 72 + m] = (_Float16)(sa[tid * 68 + m] * inv);
  }
  __syncthreads();
  {  // out = attn @ v (64x96, K=64)
    const int wr = wv >> 1, wc = wv & 1;
    v8f c[3] = {};
#pragma unroll
    for (int k0 = 0; k0 < 64; k0 += 32) {
      v16h a = load_a_frag(ah + k0, 72, wr * 16 + lrow, half);
#pragma unroll
      for (int s = 0; s < 3; ++s) {
        int dcol = wc * 48 + s * 16 + lrow;
        v16h b;
#pragma unroll
        for (int j = 0; j < 16; ++j) b[j] = vs[(k0 + half * 16 + j) * 104 + dcol];
        c[s] = wmma_f16(a, b, c[s]);
      }
    }
#pragma unroll
    for (int s = 0; s < 3; ++s) {
      int dcol = wc * 48 + s * 16 + lrow;
#pragma unroll
      for (int r = 0; r < 8; ++r) {
        int n = wr * 16 + r + 8 * half;
        out[(size_t)(wid * 64 + n) * CC + head * HD + dcol] = (_Float16)c[s][r];
      }
    }
  }
}

// ---- post-norm LayerNorm + residual (optionally emit f16 copy) --------------
__global__ __launch_bounds__(256) void k_ln_add(const float* __restrict__ y,
                                                const float* __restrict__ shortcut,
                                                const float* __restrict__ g,
                                                const float* __restrict__ b,
                                                float* __restrict__ xout,
                                                _Float16* __restrict__ hout) {
  __shared__ float red[256];
  const int row = blockIdx.x, tid = threadIdx.x;
  const float* yr = y + (size_t)row * CC;
  float v0 = yr[tid], v1 = yr[tid + 256], v2 = yr[tid + 512];
  red[tid] = v0 + v1 + v2;
  __syncthreads();
  for (int o = 128; o > 0; o >>= 1) { if (tid < o) red[tid] += red[tid + o]; __syncthreads(); }
  float mu = red[0] * (1.0f / CC);
  __syncthreads();
  float d0 = v0 - mu, d1 = v1 - mu, d2 = v2 - mu;
  red[tid] = d0 * d0 + d1 * d1 + d2 * d2;
  __syncthreads();
  for (int o = 128; o > 0; o >>= 1) { if (tid < o) red[tid] += red[tid + o]; __syncthreads(); }
  float rstd = rsqrtf(red[0] * (1.0f / CC) + 1e-5f);
  const float* sr = shortcut + (size_t)row * CC;
  float o0 = sr[tid]       + d0 * rstd * g[tid]       + b[tid];
  float o1 = sr[tid + 256] + d1 * rstd * g[tid + 256] + b[tid + 256];
  float o2 = sr[tid + 512] + d2 * rstd * g[tid + 512] + b[tid + 512];
  float* xr = xout + (size_t)row * CC;
  xr[tid] = o0; xr[tid + 256] = o1; xr[tid + 512] = o2;
  if (hout) {
    _Float16* hr = hout + (size_t)row * CC;
    hr[tid] = (_Float16)o0; hr[tid + 256] = (_Float16)o1; hr[tid + 512] = (_Float16)o2;
  }
}

extern "C" void kernel_launch(void* const* d_in, const int* in_sizes, int n_in,
                              void* d_out, int out_size, void* d_ws, size_t ws_size,
                              hipStream_t stream) {
  (void)in_sizes; (void)n_in; (void)out_size; (void)ws_size;
  const float* x      = (const float*)d_in[0];
  const float* n1g    = (const float*)d_in[1];
  const float* n1b    = (const float*)d_in[2];
  const float* n2g    = (const float*)d_in[3];
  const float* n2b    = (const float*)d_in[4];
  const float* qkv_w  = (const float*)d_in[5];
  const float* q_bias = (const float*)d_in[6];
  const float* v_bias = (const float*)d_in[7];
  const float* lsc    = (const float*)d_in[8];
  const float* cpb_w1 = (const float*)d_in[9];
  const float* cpb_b1 = (const float*)d_in[10];
  const float* cpb_w2 = (const float*)d_in[11];
  const float* proj_w = (const float*)d_in[12];
  const float* proj_b = (const float*)d_in[13];
  const float* fc1_w  = (const float*)d_in[14];
  const float* fc1_b  = (const float*)d_in[15];
  const float* fc2_w  = (const float*)d_in[16];
  const float* fc2_b  = (const float*)d_in[17];

  char* ws = (char*)d_ws;
  size_t off = 0;
  auto take = [&](size_t bytes) -> char* {
    char* p = ws + off;
    off = (off + bytes + 255) & ~(size_t)255;
    return p;
  };
  _Float16* act16  = (_Float16*)take((size_t)TOK * CC * 2);
  _Float16* qkv16  = (_Float16*)take((size_t)TOK * NQKV * 2);
  _Float16* attn16 = (_Float16*)take((size_t)TOK * CC * 2);
  _Float16* h16    = (_Float16*)take((size_t)TOK * NHID * 2);
  float*    tmp32  = (float*)take((size_t)TOK * CC * 4);
  float*    xcur   = (float*)take((size_t)TOK * CC * 4);
  float*    rpb    = (float*)take((size_t)225 * NHEAD * 4);
  float*    qkvb   = (float*)take((size_t)NQKV * 4);
  _Float16* wq16   = (_Float16*)take((size_t)NQKV * CC * 2);
  _Float16* wp16   = (_Float16*)take((size_t)CC * CC * 2);
  _Float16* w116   = (_Float16*)take((size_t)2 * NHID * CC * 2);
  _Float16* w216   = (_Float16*)take((size_t)CC * NHID * 2);

  for (int d = 0; d < 2; ++d) {
    int shift = (d & 1) ? 4 : 0;
    const float* xin = (d == 0) ? x : xcur;
    // one-time (per depth) f16 weight conversion; weights then live in L2
    k_cvt16<<<(NQKV * CC) / 256, 256, 0, stream>>>(qkv_w + (size_t)d * NQKV * CC, wq16, NQKV * CC);
    k_cvt16<<<(CC * CC) / 256, 256, 0, stream>>>(proj_w + (size_t)d * CC * CC, wp16, CC * CC);
    k_cvt16<<<(2 * NHID * CC) / 256, 256, 0, stream>>>(fc1_w + (size_t)d * 2 * NHID * CC, w116, 2 * NHID * CC);
    k_cvt16<<<(CC * NHID) / 256, 256, 0, stream>>>(fc2_w + (size_t)d * CC * NHID, w216, CC * NHID);

    k_partition<<<(TOK * CC) / 256, 256, 0, stream>>>(xin, act16, shift);
    k_qkv_bias<<<(NQKV + 255) / 256, 256, 0, stream>>>(q_bias + d * CC, v_bias + d * CC, qkvb);
    k_cpb<<<1, 256, 0, stream>>>(cpb_w1 + (size_t)d * 512 * 2, cpb_b1 + d * 512,
                                 cpb_w2 + (size_t)d * NHEAD * 512, rpb);
    k_gemm<<<dim3(NQKV / 64, TOK / 128), 256, 0, stream>>>(
        act16, wq16, qkvb, nullptr, qkv16, TOK, NQKV, CC, -1);
    k_attn<<<NWIN * NHEAD, 256, 0, stream>>>(qkv16, rpb, lsc + d * NHEAD, attn16, shift);
    k_gemm<<<dim3(CC / 64, TOK / 128), 256, 0, stream>>>(
        attn16, wp16, proj_b + d * CC, tmp32, nullptr, TOK, CC, CC, shift);
    k_ln_add<<<TOK, 256, 0, stream>>>(tmp32, xin, n1g + d * CC, n1b + d * CC, xcur, act16);
    k_gemm_glu<<<dim3(NHID / 64, TOK / 128), 256, 0, stream>>>(
        act16, w116, fc1_b + (size_t)d * 2 * NHID, h16, TOK, NHID, CC);
    k_gemm<<<dim3(CC / 64, TOK / 128), 256, 0, stream>>>(
        h16, w216, fc2_b + d * CC, tmp32, nullptr, TOK, CC, NHID, -1);
    float* xo = (d == 1) ? (float*)d_out : xcur;
    k_ln_add<<<TOK, 256, 0, stream>>>(tmp32, xcur, n2g + d * CC, n2b + d * CC, xo, nullptr);
  }
}
